// MultiEncAutoEncoder_59846074303029
// MI455X (gfx1250) — matrix operand order repaired and checked by reading, not compiled, not run on hardware
//
#include <hip/hip_runtime.h>
#include <hip/hip_bf16.h>

// Problem constants (match reference)
#define B_TOK   8192
#define DIM     768
#define NE      16
#define ED      1536
#define DICTSZ  (NE * ED)
#define KSEL    100
#define SLOTS   2

typedef __attribute__((ext_vector_type(16))) __bf16 v16bf;
typedef __attribute__((ext_vector_type(8)))  __bf16 v8bf;
typedef __attribute__((ext_vector_type(8)))  float  v8f;

union BFrag { v16bf v; v8bf h[2]; };

__device__ __forceinline__ unsigned short f2bf(float f) {
  unsigned u = __float_as_uint(f);
  unsigned r = u + 0x7FFFu + ((u >> 16) & 1u);   // round-to-nearest-even
  return (unsigned short)(r >> 16);
}

// gfx1250 async DMA: global -> LDS, 16B per active lane, tracked by ASYNCcnt.
__device__ __forceinline__ void async_ld16(unsigned ldsByteOff, const void* g) {
  asm volatile("global_load_async_to_lds_b128 %0, %1, off"
               :: "v"(ldsByteOff), "v"(g) : "memory");
}
__device__ __forceinline__ void wait_async0() {
  asm volatile("s_wait_asynccnt 0x0" ::: "memory");
}

// ---------------------------------------------------------------- converts --
__global__ void convert_w_kernel(const float* __restrict__ W,
                                 unsigned short* __restrict__ Wb, int n) {
  int i = (blockIdx.x * 256 + threadIdx.x) * 4;
  if (i + 3 < n) {
    float4 f = *(const float4*)(W + i);
    Wb[i + 0] = f2bf(f.x);
    Wb[i + 1] = f2bf(f.y);
    Wb[i + 2] = f2bf(f.z);
    Wb[i + 3] = f2bf(f.w);
  }
}

__global__ void convert_x_kernel(const float* __restrict__ x,
                                 const float* __restrict__ bdec,
                                 unsigned short* __restrict__ xb,
                                 int* __restrict__ counts) {
  int gid = blockIdx.x * 256 + threadIdx.x;
  if (gid < NE) counts[gid] = 0;                 // zero routing counters
  int i = gid * 4;
  if (i + 3 < B_TOK * DIM) {
    float4 f = *(const float4*)(x + i);
    int d = i % DIM;                             // 4 | 768 -> chunk stays in row
    xb[i + 0] = f2bf(f.x - bdec[d + 0]);
    xb[i + 1] = f2bf(f.y - bdec[d + 1]);
    xb[i + 2] = f2bf(f.z - bdec[d + 2]);
    xb[i + 3] = f2bf(f.w - bdec[d + 3]);
  }
}

// ------------------------------------------------------------- gate + route --
__global__ __launch_bounds__(256)
void gate_route_kernel(const float* __restrict__ x, const float* __restrict__ gW,
                       const float* __restrict__ gb, const float* __restrict__ bgate,
                       int* __restrict__ counts, unsigned* __restrict__ routed,
                       float* __restrict__ routedw, int* __restrict__ topE) {
  int wid  = threadIdx.x >> 5;    // wave32
  int lane = threadIdx.x & 31;
  int tok  = blockIdx.x * 8 + wid;
  if (tok >= B_TOK) return;

  float part[NE];
#pragma unroll
  for (int e = 0; e < NE; ++e) part[e] = 0.f;
  const float* xr = x + (size_t)tok * DIM;
  for (int d = lane; d < DIM; d += 32) {
    float xv = xr[d] - bgate[d];
#pragma unroll
    for (int e = 0; e < NE; ++e) part[e] += xv * gW[e * DIM + d];
  }
#pragma unroll
  for (int e = 0; e < NE; ++e) {
    float v = part[e];
    for (int off = 16; off > 0; off >>= 1) v += __shfl_xor(v, off, 32);
    part[e] = v;
  }
  if (lane == 0) {
    float m = -1e30f;
#pragma unroll
    for (int e = 0; e < NE; ++e) { part[e] += gb[e]; m = fmaxf(m, part[e]); }
    float p[NE]; float s = 0.f;
#pragma unroll
    for (int e = 0; e < NE; ++e) { p[e] = __expf(part[e] - m); s += p[e]; }
    float inv = 1.f / s;
#pragma unroll
    for (int e = 0; e < NE; ++e) p[e] *= inv;
    int i0 = 0;
#pragma unroll
    for (int e = 1; e < NE; ++e) if (p[e] > p[i0]) i0 = e;
    int i1 = (i0 == 0) ? 1 : 0;
#pragma unroll
    for (int e = 0; e < NE; ++e) if (e != i0 && p[e] > p[i1]) i1 = e;
    // reference: softmax over the two top PROBABILITIES
    float w0 = 1.f / (1.f + __expf(p[i1] - p[i0]));
    float w1 = 1.f - w0;
    topE[tok * 2 + 0] = i0;
    topE[tok * 2 + 1] = i1;
    int q0 = atomicAdd(&counts[i0], 1);
    routed[i0 * B_TOK + q0]  = (unsigned)(tok * 2 + 0);
    routedw[i0 * B_TOK + q0] = w0;
    int q1 = atomicAdd(&counts[i1], 1);
    routed[i1 * B_TOK + q1]  = (unsigned)(tok * 2 + 1);
    routedw[i1 * B_TOK + q1] = w1;
  }
}

// ------------------------------------------------- routed expert WMMA GEMM --
// Block tile 64(M) x 128(N), K-step 32; 8 waves, each 32x32 via 2x2 WMMA.
// Double-buffered LDS staging via GLOBAL_LOAD_ASYNC_TO_LDS_B128 (ASYNCcnt),
// one s_wait_asynccnt + one barrier per K-step; DMA(k+1) overlaps WMMA(k).
__global__ __launch_bounds__(256)
void moe_gemm_kernel(const unsigned short* __restrict__ xb,   // [B_TOK][DIM] bf16(x-b_dec)
                     const unsigned short* __restrict__ Wb,   // [NE][ED][DIM] bf16
                     const float* __restrict__ expert_b,      // [NE][ED]
                     const int* __restrict__ counts,
                     const unsigned* __restrict__ routed,
                     const float* __restrict__ routedw,
                     float* __restrict__ fsel)                // [B_TOK][2][ED]
{
  const int e   = blockIdx.z;
  const int cnt = counts[e];
  const int m0  = blockIdx.y * 64;
  if (m0 >= cnt) return;                         // early-exit empty tiles
  const int n0  = blockIdx.x * 128;

  __shared__ __align__(16) unsigned short At[2][64 * 40];    // +8 bf16 row pad
  __shared__ __align__(16) unsigned short Bt[2][128 * 40];

  const int tid  = threadIdx.x;
  const int lane = tid & 31;
  const int wid  = tid >> 5;
  const int mBase = (wid & 1) * 32;
  const int nBase = (wid >> 1) * 32;

  v8f acc[2][2];
#pragma unroll
  for (int i = 0; i < 2; ++i)
#pragma unroll
    for (int j = 0; j < 2; ++j)
#pragma unroll
      for (int r = 0; r < 8; ++r) acc[i][j][r] = 0.f;

  // Staging map: A = 64 rows x 32 cols (4 thr/row, 16B each);
  //              B = 128 rows x 32 cols (two 256-thread passes).
  const int arow = tid >> 2;
  const int acol = (tid & 3) * 8;
  const bool rowValid = (m0 + arow) < cnt;
  int tokA = 0;
  if (rowValid) tokA = (int)(routed[e * B_TOK + m0 + arow] >> 1);
  const unsigned short* aSrc = xb + (size_t)tokA * DIM + acol;
  const unsigned short* wB   = Wb + (size_t)e * ED * DIM;

  // LDS byte offsets (flat shared addr low 32 bits == wave-relative LDS offset)
  const unsigned AtBase = (unsigned)(uintptr_t)(void*)&At[0][0];
  const unsigned BtBase = (unsigned)(uintptr_t)(void*)&Bt[0][0];
  const unsigned ldsA   = AtBase + (unsigned)(arow * 40 + acol) * 2;
  const int brow0 = tid >> 1 >> 1;               // tid>>2
  const int brow1 = (tid + 256) >> 2;
  const int bcol0 = (tid & 3) * 8;
  const int bcol1 = ((tid + 256) & 3) * 8;
  const unsigned ldsB0 = BtBase + (unsigned)(brow0 * 40 + bcol0) * 2;
  const unsigned ldsB1 = BtBase + (unsigned)(brow1 * 40 + bcol1) * 2;

  auto issueTile = [&](int buf, int k0) {
    if (rowValid)
      async_ld16(ldsA + (unsigned)buf * (64 * 40 * 2), aSrc + k0);
    async_ld16(ldsB0 + (unsigned)buf * (128 * 40 * 2),
               wB + (size_t)(n0 + brow0) * DIM + k0 + bcol0);
    async_ld16(ldsB1 + (unsigned)buf * (128 * 40 * 2),
               wB + (size_t)(n0 + brow1) * DIM + k0 + bcol1);
  };

  issueTile(0, 0);                               // prologue

  const int hsel = lane >> 4;                    // CDNA5 16-bit WMMA lane halves
  const int l15  = lane & 15;
  const int KSTEPS = DIM / 32;

  for (int ks = 0; ks < KSTEPS; ++ks) {
    const int cur = ks & 1;
    wait_async0();                               // my DMA into buf `cur` done
    __syncthreads();                             // everyone's DMA visible; all
                                                 // reads of buf cur^1 finished
    if (ks + 1 < KSTEPS) issueTile(cur ^ 1, (ks + 1) * 32);

    const unsigned short* Ab = At[cur];
    const unsigned short* Bb = Bt[cur];
    BFrag a[2], b[2];
#pragma unroll
    for (int i = 0; i < 2; ++i) {
      const unsigned short* ap = Ab + (mBase + i * 16 + l15) * 40;
      a[i].h[0] = *(const v8bf*)(ap + hsel * 8);        // K = h*8 .. h*8+7
      a[i].h[1] = *(const v8bf*)(ap + 16 + hsel * 8);   // K = 16+h*8 ..
    }
#pragma unroll
    for (int j = 0; j < 2; ++j) {
      const unsigned short* bp = Bb + (nBase + j * 16 + l15) * 40 + hsel * 8;
      b[j].h[0] = *(const v8bf*)(bp);                   // K = h*8 .. h*8+7
      b[j].h[1] = *(const v8bf*)(bp + 16);              // K = 16+h*8 .. +7
    }
#pragma unroll
    for (int i = 0; i < 2; ++i)
#pragma unroll
      for (int j = 0; j < 2; ++j)
        acc[i][j] = __builtin_amdgcn_wmma_f32_16x16x32_bf16(
            false, a[i].v, false, b[j].v, (short)0, acc[i][j], false, false);
  }

  // Epilogue: +expert_b, relu, * gate weight, scatter to f_sel[token][slot][:]
  const int mOff = (lane < 16) ? 0 : 8;          // f32 C/D layout
#pragma unroll
  for (int i = 0; i < 2; ++i) {
#pragma unroll
    for (int r = 0; r < 8; ++r) {
      int gRow = m0 + mBase + i * 16 + mOff + r;
      if (gRow >= cnt) continue;
      unsigned ent = routed[e * B_TOK + gRow];
      int tok  = (int)(ent >> 1);
      int slot = (int)(ent & 1u);
      float w  = routedw[e * B_TOK + gRow];
#pragma unroll
      for (int j = 0; j < 2; ++j) {
        int col = n0 + nBase + j * 16 + l15;
        float v = acc[i][j][r] + expert_b[e * ED + col];
        v = fmaxf(v, 0.f) * w;
        fsel[((size_t)tok * SLOTS + slot) * ED + col] = v;
      }
    }
  }
}

// ------------------------------------------------ top-K(100) + L2 decode ----
__global__ __launch_bounds__(256)
void topk_decode_kernel(const float* __restrict__ fsel, const int* __restrict__ topE,
                        const float* __restrict__ decoder, const float* __restrict__ bdec,
                        float* __restrict__ out) {
  __shared__ float fv[SLOTS * ED];               // 3072 candidate activations
  __shared__ unsigned hist[256];
  __shared__ float sa[128];
  __shared__ int   sd[128];
  __shared__ int sBin, sAbove, sNeed, sTotal, cntHi, cntTie;

  const int tok = blockIdx.x;
  const int tid = threadIdx.x;
  const float* frow = fsel + (size_t)tok * (SLOTS * ED);

  for (int p = tid; p < SLOTS * ED; p += 256) fv[p] = frow[p];
  hist[tid] = 0;
  if (tid == 0) { cntHi = 0; cntTie = 0; }
  __syncthreads();

  // all values >= 0 -> uint order == float order; bucket by exponent byte
  for (int p = tid; p < SLOTS * ED; p += 256)
    atomicAdd(&hist[(__float_as_uint(fv[p]) >> 23) & 255u], 1u);
  __syncthreads();

  if (tid == 0) {
    int t = 255, cum = 0;
    while (t > 0 && cum + (int)hist[t] < KSEL) { cum += (int)hist[t]; --t; }
    int need = KSEL - cum;
    int h = (int)hist[t];
    if (need > h) need = h;
    sBin = t; sAbove = cum; sNeed = need; sTotal = cum + need;
  }
  __syncthreads();

  const int e0 = topE[tok * 2 + 0];
  const int e1 = topE[tok * 2 + 1];
  const int tThr = sBin, need = sNeed, above = sAbove;

  for (int p = tid; p < SLOTS * ED; p += 256) {
    float v = fv[p];
    int bin = (int)((__float_as_uint(v) >> 23) & 255u);
    if (bin > tThr) {
      int q = atomicAdd(&cntHi, 1);
      int j = (p < ED) ? p : (p - ED);
      sa[q] = v; sd[q] = ((p < ED) ? e0 : e1) * ED + j;
    } else if (bin == tThr) {
      int q = atomicAdd(&cntTie, 1);
      if (q < need) {
        int w = above + q;
        int j = (p < ED) ? p : (p - ED);
        sa[w] = v; sd[w] = ((p < ED) ? e0 : e1) * ED + j;
      }
    }
  }
  __syncthreads();

  // x_hat = sum acts * decoder[idx] + b_dec ; decoder (75.5MB) is L2-resident
  const int total = sTotal;
  const int c0 = tid * 3;                        // 768 = 256 * 3
  float a0 = 0.f, a1 = 0.f, a2 = 0.f;
  for (int s = 0; s < total; ++s) {
    if (s + 4 < total)                           // hide L2 latency on gather
      __builtin_prefetch(decoder + (size_t)sd[s + 4] * DIM + c0, 0, 1);
    float a = sa[s];
    const float* dr = decoder + (size_t)sd[s] * DIM + c0;
    a0 += a * dr[0];
    a1 += a * dr[1];
    a2 += a * dr[2];
  }
  float* orow = out + (size_t)tok * DIM + c0;
  orow[0] = a0 + bdec[c0 + 0];
  orow[1] = a1 + bdec[c0 + 1];
  orow[2] = a2 + bdec[c0 + 2];
}

// ---------------------------------------------------------------------------
extern "C" void kernel_launch(void* const* d_in, const int* in_sizes, int n_in,
                              void* d_out, int out_size, void* d_ws, size_t ws_size,
                              hipStream_t stream) {
  const float* x       = (const float*)d_in[0];
  const float* gate_W  = (const float*)d_in[1];
  const float* gate_b  = (const float*)d_in[2];
  const float* b_gate  = (const float*)d_in[3];
  const float* b_dec   = (const float*)d_in[4];
  const float* exp_W   = (const float*)d_in[5];
  const float* exp_b   = (const float*)d_in[6];
  const float* decoder = (const float*)d_in[7];
  float* out = (float*)d_out;
  (void)in_sizes; (void)n_in; (void)out_size; (void)ws_size;

  char* ws = (char*)d_ws;
  size_t off = 0;
  auto alloc = [&](size_t bytes) -> void* {
    void* p = (void*)(ws + off);
    off += (bytes + 255) & ~(size_t)255;
    return p;
  };
  unsigned short* Wb   = (unsigned short*)alloc((size_t)NE * ED * DIM * 2);  // 37.7 MB
  unsigned short* xb   = (unsigned short*)alloc((size_t)B_TOK * DIM * 2);    // 12.6 MB
  float* fsel          = (float*)alloc((size_t)B_TOK * SLOTS * ED * 4);      // 100.7 MB
  int* counts          = (int*)alloc((size_t)NE * 4);
  unsigned* routed     = (unsigned*)alloc((size_t)NE * B_TOK * 4);
  float* routedw       = (float*)alloc((size_t)NE * B_TOK * 4);
  int* topE            = (int*)alloc((size_t)B_TOK * 2 * 4);

  const int nW = NE * ED * DIM;
  convert_w_kernel<<<nW / (256 * 4), 256, 0, stream>>>(exp_W, Wb, nW);
  convert_x_kernel<<<(B_TOK * DIM) / (256 * 4), 256, 0, stream>>>(x, b_dec, xb, counts);
  gate_route_kernel<<<B_TOK / 8, 256, 0, stream>>>(x, gate_W, gate_b, b_gate,
                                                   counts, routed, routedw, topE);
  moe_gemm_kernel<<<dim3(ED / 128, B_TOK / 64, NE), 256, 0, stream>>>(
      xb, Wb, exp_b, counts, routed, routedw, fsel);
  topk_decode_kernel<<<B_TOK, 256, 0, stream>>>(fsel, topE, decoder, b_dec, out);
}
